// BaselineNet_75161927680493
// MI455X (gfx1250) — matrix operand order repaired and checked by reading, not compiled, not run on hardware
//
#include <hip/hip_runtime.h>

// GCN 3-layer forward for MI455X (gfx1250, wave32, WMMA).
// Pipeline per layer:  Hs = dinv .* (A @ W)   (bf16 WMMA, f32 accum)
//                      acc = Hs ; acc[d] += Hs[s] for every edge (f32 atomics, L2-resident)
//                      out = act(dinv .* acc + b)
// deg/dinv computed once (depends only on dst + self loops).

typedef __attribute__((ext_vector_type(16))) __bf16 v16bf;
typedef __attribute__((ext_vector_type(8)))  float  v8f;

#define N_NODES 50000
#define N_EDGES 800000
#define F_IN  256
#define F_MID 128
#define F_OUT 40

__device__ __forceinline__ unsigned short f32_to_bf16(float f) {
  unsigned u = __float_as_uint(f);
  u += 0x7FFFu + ((u >> 16) & 1u);          // round-to-nearest-even
  return (unsigned short)(u >> 16);
}

// ---------------- degree / normalization ----------------
__global__ void deg_init_kernel(float* __restrict__ deg, int n) {
  int i = blockIdx.x * blockDim.x + threadIdx.x;
  if (i < n) deg[i] = 1.0f;                 // self loop
}

__global__ void deg_accum_kernel(const long long* __restrict__ dst, float* __restrict__ deg, int e) {
  int i = blockIdx.x * blockDim.x + threadIdx.x;
  if (i < e) atomicAdd(&deg[(int)dst[i]], 1.0f);
}

__global__ void dinv_kernel(const float* __restrict__ deg, float* __restrict__ dinv, int n) {
  int i = blockIdx.x * blockDim.x + threadIdx.x;
  if (i < n) dinv[i] = rsqrtf(deg[i]);      // deg >= 1 always
}

// ---------------- f32 -> bf16 activation convert ----------------
__global__ void cvt_bf16_kernel(const float* __restrict__ in, unsigned short* __restrict__ out,
                                size_t cnt) {
  size_t i = (size_t)blockIdx.x * blockDim.x + threadIdx.x;
  if (i < cnt) out[i] = f32_to_bf16(in[i]);
}

// ---------------- pack W [K,N] f32 (row-major) into WMMA B-fragment layout ----------------
// Bp index: (((nt*KT + kt)*32 + lane)*16 + j)
//   col = nt*16 + (lane&15);  k = kt*32 + (lane>>4)*8 + (j&7) + (j>>3)*16   (cols >= N zero-padded)
__global__ void pack_b_kernel(const float* __restrict__ W, unsigned short* __restrict__ Bp,
                              int K, int N, int NT) {
  int idx = blockIdx.x * blockDim.x + threadIdx.x;
  int KT = K / 32;
  int total = NT * KT * 32 * 16;
  if (idx >= total) return;
  int j    = idx & 15;
  int lane = (idx >> 4) & 31;
  int kt   = (idx >> 9) % KT;
  int nt   = (idx >> 9) / KT;
  int col  = nt * 16 + (lane & 15);
  int k    = kt * 32 + (lane >> 4) * 8 + (j & 7) + (j >> 3) * 16;
  float v  = (col < N) ? W[(size_t)k * N + col] : 0.0f;
  Bp[idx]  = f32_to_bf16(v);
}

// ---------------- WMMA GEMM: Hs = dinv[row] * (A @ W); acc = Hs ----------------
// One wave per 16-row strip; NT column tiles held in accumulators; K stepped by 32.
// A: row-major bf16 [M,K].  A-fragment per documented 16-bit 16x32 layout:
//   lane<16: row=lane,   K = {kb..kb+7} U {kb+16..kb+23} with kb = kt*32
//   lane>=16: row=lane-16, same with kb = kt*32 + 8
template <int K, int NT, int NOUT>
__global__ void gemm_wmma_kernel(const unsigned short* __restrict__ A,
                                 const unsigned short* __restrict__ Bp,
                                 const float* __restrict__ dinv,
                                 float* __restrict__ Hs, float* __restrict__ acc,
                                 int mtiles) {
  int wid = (int)((blockIdx.x * blockDim.x + threadIdx.x) >> 5);
  if (wid >= mtiles) return;                 // whole-wave exit: EXEC stays all-1s for WMMA
  int lane = threadIdx.x & 31;
  int row  = lane & 15;
  int half = lane >> 4;
  const int KT = K / 32;

  v8f c[NT] = {};
  size_t arow = (size_t)(wid * 16 + row) * K;

#pragma unroll
  for (int kt = 0; kt < KT; ++kt) {
    union { uint4 u[2]; v16bf v; } af;
    const unsigned short* ap = A + arow + kt * 32 + half * 8;
    af.u[0] = *(const uint4*)(ap);
    af.u[1] = *(const uint4*)(ap + 16);
#pragma unroll
    for (int nt = 0; nt < NT; ++nt) {
      union { uint4 u[2]; v16bf v; } bf_;
      const unsigned short* bp = Bp + ((size_t)(nt * KT + kt) * 32 + lane) * 16;
      bf_.u[0] = *(const uint4*)(bp);
      bf_.u[1] = *(const uint4*)(bp + 8);
      c[nt] = __builtin_amdgcn_wmma_f32_16x16x32_bf16(
          false, af.v, false, bf_.v, (short)0, c[nt], false, false);
    }
  }

  // Epilogue: C/D layout — lane<16: M=r, N=lane; lane>=16: M=8+r, N=lane-16.
#pragma unroll
  for (int nt = 0; nt < NT; ++nt) {
#pragma unroll
    for (int r = 0; r < 8; ++r) {
      int gm  = wid * 16 + r + half * 8;
      int col = nt * 16 + (lane & 15);
      if (col < NOUT) {
        float v = c[nt][r] * dinv[gm];
        Hs[(size_t)gm * NOUT + col]  = v;   // read-only copy for the scatter
        acc[(size_t)gm * NOUT + col] = v;   // accumulator pre-seeded with self-loop term
      }
    }
  }
}

// ---------------- edge scatter: acc[d] += Hs[s] ----------------
__global__ void scatter_kernel(const float* __restrict__ Hs, float* __restrict__ acc,
                               const long long* __restrict__ src,
                               const long long* __restrict__ dst,
                               int e, int F) {
  int wave = (int)((blockIdx.x * blockDim.x + threadIdx.x) >> 5);
  int lane = threadIdx.x & 31;
  if (wave >= e) return;
  int s = (int)src[wave];                    // scalar (uniform) per wave
  int d = (int)dst[wave];
  const float* hp = Hs  + (size_t)s * F;
  float*       ap = acc + (size_t)d * F;
  for (int f = lane; f < F; f += 32)
    atomicAdd(ap + f, hp[f]);
}

// ---------------- finalize: out = relu(dinv*acc + b) -> bf16 activation ----------------
__global__ void finalize_relu_kernel(const float* __restrict__ acc, const float* __restrict__ dinv,
                                     const float* __restrict__ bias,
                                     unsigned short* __restrict__ act, int n, int F) {
  size_t i = (size_t)blockIdx.x * blockDim.x + threadIdx.x;
  if (i >= (size_t)n * F) return;
  int v = (int)(i / F), f = (int)(i % F);
  float val = fmaxf(dinv[v] * acc[i] + bias[f], 0.0f);
  act[i] = f32_to_bf16(val);
}

// ---------------- finalize (last layer, no relu): f32 logits ----------------
__global__ void finalize_lin_kernel(const float* __restrict__ acc, const float* __restrict__ dinv,
                                    const float* __restrict__ bias,
                                    float* __restrict__ logits, int n, int F) {
  size_t i = (size_t)blockIdx.x * blockDim.x + threadIdx.x;
  if (i >= (size_t)n * F) return;
  int v = (int)(i / F), f = (int)(i % F);
  logits[i] = dinv[v] * acc[i] + bias[f];
}

// ---------------- log-softmax over C=40 classes, one wave per node ----------------
__global__ void log_softmax_kernel(const float* __restrict__ logits, float* __restrict__ out,
                                   int n, int C) {
  int node = (int)((blockIdx.x * blockDim.x + threadIdx.x) >> 5);
  int lane = threadIdx.x & 31;
  if (node >= n) return;
  const float* p = logits + (size_t)node * C;
  float v0 = (lane      < C) ? p[lane]      : -INFINITY;
  float v1 = (lane + 32 < C) ? p[lane + 32] : -INFINITY;
  float m = fmaxf(v0, v1);
#pragma unroll
  for (int off = 16; off > 0; off >>= 1) m = fmaxf(m, __shfl_xor(m, off, 32));
  float e = ((lane < C) ? __expf(v0 - m) : 0.0f) + ((lane + 32 < C) ? __expf(v1 - m) : 0.0f);
#pragma unroll
  for (int off = 16; off > 0; off >>= 1) e += __shfl_xor(e, off, 32);
  float lse = m + __logf(e);
  if (lane      < C) out[(size_t)node * C + lane]      = v0 - lse;
  if (lane + 32 < C) out[(size_t)node * C + lane + 32] = v1 - lse;
}

// ===================================================================================

extern "C" void kernel_launch(void* const* d_in, const int* in_sizes, int n_in,
                              void* d_out, int out_size, void* d_ws, size_t ws_size,
                              hipStream_t stream) {
  const float*     x     = (const float*)d_in[0];
  const long long* eidx  = (const long long*)d_in[1];   // [2, E] int64 per reference
  const float*     W_in  = (const float*)d_in[2];
  const float*     b_in  = (const float*)d_in[3];
  const float*     W_mid = (const float*)d_in[4];
  const float*     b_mid = (const float*)d_in[5];
  const float*     W_out = (const float*)d_in[6];
  const float*     b_out = (const float*)d_in[7];
  float*           out   = (float*)d_out;

  const int n = in_sizes[0] / F_IN;          // 50000 (multiple of 16)
  const int e = in_sizes[1] / 2;             // 800000
  const long long* src = eidx;
  const long long* dst = eidx + e;

  // ---- carve workspace (everything read is fully written first) ----
  size_t off = 0;
  auto carve = [&](size_t bytes) -> void* {
    void* p = (char*)d_ws + off;
    off += (bytes + 255) & ~(size_t)255;
    return p;
  };
  float*          deg    = (float*)carve((size_t)n * 4);
  float*          dinv   = (float*)carve((size_t)n * 4);
  unsigned short* Xb     = (unsigned short*)carve((size_t)n * F_IN * 2);   // bf16 input L1
  unsigned short* Act    = (unsigned short*)carve((size_t)n * F_MID * 2);  // bf16 input L2/L3
  float*          Hs     = (float*)carve((size_t)n * F_MID * 4);
  float*          accb   = (float*)carve((size_t)n * F_MID * 4);
  unsigned short* Bp     = (unsigned short*)carve((size_t)8 * (F_IN / 32) * 32 * 16 * 2);
  float*          logits = (float*)carve((size_t)n * F_OUT * 4);
  (void)ws_size;

  const int T = 256;
  const int mtiles = n / 16;
  const int gemm_blocks    = (mtiles + 7) / 8;          // 8 waves / block
  const int scatter_blocks = (e + 7) / 8;               // 1 wave / edge

  // ---- normalization (once) ----
  deg_init_kernel<<<(n + T - 1) / T, T, 0, stream>>>(deg, n);
  deg_accum_kernel<<<(e + T - 1) / T, T, 0, stream>>>(dst, deg, e);
  dinv_kernel<<<(n + T - 1) / T, T, 0, stream>>>(deg, dinv, n);

  // ---- layer 1: F_IN -> F_MID, relu ----
  cvt_bf16_kernel<<<(int)(((size_t)n * F_IN + T - 1) / T), T, 0, stream>>>(x, Xb, (size_t)n * F_IN);
  {
    int total = 8 * (F_IN / 32) * 32 * 16;
    pack_b_kernel<<<(total + T - 1) / T, T, 0, stream>>>(W_in, Bp, F_IN, F_MID, 8);
  }
  gemm_wmma_kernel<F_IN, 8, F_MID><<<gemm_blocks, T, 0, stream>>>(Xb, Bp, dinv, Hs, accb, mtiles);
  scatter_kernel<<<scatter_blocks, T, 0, stream>>>(Hs, accb, src, dst, e, F_MID);
  finalize_relu_kernel<<<(int)(((size_t)n * F_MID + T - 1) / T), T, 0, stream>>>(
      accb, dinv, b_in, Act, n, F_MID);

  // ---- layer 2: F_MID -> F_MID, relu ----
  {
    int total = 8 * (F_MID / 32) * 32 * 16;
    pack_b_kernel<<<(total + T - 1) / T, T, 0, stream>>>(W_mid, Bp, F_MID, F_MID, 8);
  }
  gemm_wmma_kernel<F_MID, 8, F_MID><<<gemm_blocks, T, 0, stream>>>(Act, Bp, dinv, Hs, accb, mtiles);
  scatter_kernel<<<scatter_blocks, T, 0, stream>>>(Hs, accb, src, dst, e, F_MID);
  finalize_relu_kernel<<<(int)(((size_t)n * F_MID + T - 1) / T), T, 0, stream>>>(
      accb, dinv, b_mid, Act, n, F_MID);

  // ---- layer 3: F_MID -> F_OUT(=40, padded to 48 cols), log-softmax ----
  {
    int total = 3 * (F_MID / 32) * 32 * 16;
    pack_b_kernel<<<(total + T - 1) / T, T, 0, stream>>>(W_out, Bp, F_MID, F_OUT, 3);
  }
  gemm_wmma_kernel<F_MID, 3, F_OUT><<<gemm_blocks, T, 0, stream>>>(Act, Bp, dinv, Hs, accb, mtiles);
  scatter_kernel<<<scatter_blocks, T, 0, stream>>>(Hs, accb, src, dst, e, F_OUT);
  finalize_lin_kernel<<<(int)(((size_t)n * F_OUT + T - 1) / T), T, 0, stream>>>(
      accb, dinv, b_out, logits, n, F_OUT);
  log_softmax_kernel<<<(n + 7) / 8, T, 0, stream>>>(logits, out, n, F_OUT);
}
